// ProjectionLayer_35759897706836
// MI455X (gfx1250) — matrix operand order repaired and checked by reading, not compiled, not run on hardware
//
#include <hip/hip_runtime.h>
#include <hip/hip_bf16.h>
#include <math.h>

typedef __attribute__((ext_vector_type(16))) __bf16 v16bf;
typedef __attribute__((ext_vector_type(8)))  __bf16 v8bf;
typedef __attribute__((ext_vector_type(8)))  float  v8f;

#define D_DIM 1024
#define H_DIM 512
#define O_DIM 256
#define BATCH 64
#define KSEQ  512
#define MROWS (BATCH * KSEQ)   // 32768
#define MTILE 32
#define LDPAD 40               // padded LDS row stride (bf16 elems), 80B -> conflict-free, 16B aligned
#define NSTEP (D_DIM / 32)     // 32 k-steps

__device__ __forceinline__ v16bf join16(v8bf lo, v8bf hi) {
  return __builtin_shufflevector(lo, hi, 0,1,2,3,4,5,6,7,8,9,10,11,12,13,14,15);
}

// Async global->LDS copy of one 32-k slice of Wt (bf16 [H][D]) into a padded LDS buffer.
// 512 rows x 32 k bf16 = 2048 x 16B chunks; 256 threads x 8 -> 8 async instrs per wave.
__device__ __forceinline__ void issue_B_async(const __bf16* __restrict__ Wt, __bf16* sBbuf, int ks) {
  const int t = threadIdx.x;
#pragma unroll
  for (int i = 0; i < 8; ++i) {
    int idx = t + i * 256;                 // 0..2047
    int n = idx >> 2, c = (idx & 3) * 8;   // row n, 8-elem (16B) chunk c
    unsigned lds = (unsigned)(uintptr_t)(sBbuf + n * LDPAD + c);
    const void* g = (const void*)(Wt + (size_t)n * D_DIM + ks + c);
    asm volatile("global_load_async_to_lds_b128 %0, %1, off"
                 :: "v"(lds), "v"(g) : "memory");
  }
}

// Convert + transpose W[d][h] (fp32, DxH) -> Wt[h][d] (bf16, HxD) for both big weights.
__global__ void prep_weights(const float* __restrict__ Was, const float* __restrict__ Wc1,
                             __bf16* __restrict__ WasT, __bf16* __restrict__ Wc1T) {
  int tid = blockIdx.x * blockDim.x + threadIdx.x;   // 0 .. D*H-1, tid = h*D + d
  int d = tid % D_DIM;
  int h = tid / D_DIM;
  WasT[tid] = (__bf16)Was[d * H_DIM + h];
  Wc1T[tid] = (__bf16)Wc1[d * H_DIM + h];
}

// Shared WMMA mainloop: block computes a 32 x 512 f32 accumulator of x[m0:m0+32,:] @ W
// (W transposed bf16 [H][D]). 8 waves; wave w owns m_sub = w>>2 (16 rows) and 8 N-tiles.
// B tiles double-buffered and staged with async global->LDS copies one step ahead.
__device__ __forceinline__ void gemm_tile_mainloop(
    const float* __restrict__ x, const __bf16* __restrict__ Wt,
    int m0, int lane, int m_sub, int n_group,
    __bf16* sA, __bf16* sB0, __bf16* sB1, v8f acc[8])
{
  const int t      = threadIdx.x;
  const int arow   = t >> 3;             // 0..31 (A tile row to stage)
  const int acol   = (t & 7) * 4;        // 4 consecutive k per thread
  const int klo    = (lane < 16) ? 0 : 8;    // A-frag K group base (elems 0-7)
  const int kloB   = (lane < 16) ? 0 : 8;    // B-frag K base
  const int arow_f = m_sub * 16 + (lane & 15);

  issue_B_async(Wt, sB0, 0);             // prefill buffer 0

  for (int s = 0; s < NSTEP; ++s) {
    const int ks = s * 32;
    __bf16* sBc = (s & 1) ? sB1 : sB0;   // compute buffer
    __bf16* sBn = (s & 1) ? sB0 : sB1;   // next (being filled)
    __syncthreads();                     // all waves done reading sBn (prev compute) and sA
    if (s + 1 < NSTEP) issue_B_async(Wt, sBn, ks + 32);
    // stage A synchronously: 32 rows x 32 k, fp32 -> bf16
    {
      const float4 v = *(const float4*)(x + (size_t)(m0 + arow) * D_DIM + ks + acol);
      __bf16* p = sA + arow * LDPAD + acol;
      p[0] = (__bf16)v.x; p[1] = (__bf16)v.y; p[2] = (__bf16)v.z; p[3] = (__bf16)v.w;
    }
    // ASYNCcnt retires in order: waiting <=8 guarantees the batch for step s is done.
    if (s + 1 < NSTEP) asm volatile("s_wait_asynccnt 0x8" ::: "memory");
    else               asm volatile("s_wait_asynccnt 0x0" ::: "memory");
    __syncthreads();                     // A stores + every wave's B copies visible
    // A fragment: lanes 0-15 hold K 0-7 & 16-23, lanes 16-31 hold K 8-15 & 24-31 of row lane%16
    v8bf alo = *(const v8bf*)(sA + arow_f * LDPAD + klo);
    v8bf ahi = *(const v8bf*)(sA + arow_f * LDPAD + klo + 16);
    v16bf a = join16(alo, ahi);
#pragma unroll
    for (int j = 0; j < 8; ++j) {
      int n = (n_group * 8 + j) * 16 + (lane & 15);
      v8bf blo = *(const v8bf*)(sBc + n * LDPAD + kloB);
      v8bf bhi = *(const v8bf*)(sBc + n * LDPAD + kloB + 16);
      v16bf b = join16(blo, bhi);
      acc[j] = __builtin_amdgcn_wmma_f32_16x16x32_bf16(
          false, a, false, b, (short)0, acc[j], false, false);
    }
  }
}

// scores[m] = tanh(x[m]@W_as + b_as) . W_af + b_af
__global__ void __launch_bounds__(256)
scores_kernel(const float* __restrict__ x, const __bf16* __restrict__ WasT,
              const float* __restrict__ b_as, const float* __restrict__ W_af,
              const float* __restrict__ b_af, float* __restrict__ scores)
{
  __shared__ __bf16 sA[MTILE * LDPAD];
  __shared__ __bf16 sB[2][H_DIM * LDPAD];
  __shared__ float  sSc[MTILE];
  const int t = threadIdx.x, lane = t & 31, wave = t >> 5;
  const int m_sub = wave >> 2, n_group = wave & 3;
  const int m0 = blockIdx.x * MTILE;

  v8f acc[8];
  v8f z = {0.f,0.f,0.f,0.f,0.f,0.f,0.f,0.f};
#pragma unroll
  for (int j = 0; j < 8; ++j) acc[j] = z;
  if (t < MTILE) sSc[t] = 0.f;

  gemm_tile_mainloop(x, WasT, m0, lane, m_sub, n_group, sA, sB[0], sB[1], acc);

  const int rh = (lane < 16) ? 0 : 8;    // C/D layout: VGPR v -> row v (lanes 0-15), v+8 (16-31)
  float pr[8];
#pragma unroll
  for (int v = 0; v < 8; ++v) pr[v] = 0.f;
#pragma unroll
  for (int j = 0; j < 8; ++j) {
    int col = (n_group * 8 + j) * 16 + (lane & 15);
    float bias = b_as[col], waf = W_af[col];
#pragma unroll
    for (int v = 0; v < 8; ++v)
      pr[v] += tanhf(acc[j][v] + bias) * waf;
  }
  __syncthreads();
#pragma unroll
  for (int v = 0; v < 8; ++v)
    atomicAdd(&sSc[m_sub * 16 + rh + v], pr[v]);
  __syncthreads();
  if (t < MTILE) scores[m0 + t] = sSc[t] + b_af[0];
}

__global__ void __launch_bounds__(512)
softmax_kernel(const float* __restrict__ scores, float* __restrict__ attn)
{
  __shared__ float red[512];
  const int b = blockIdx.x, t = threadIdx.x;
  float s = scores[b * KSEQ + t];
  red[t] = s; __syncthreads();
  for (int off = 256; off > 0; off >>= 1) {
    if (t < off) red[t] = fmaxf(red[t], red[t + off]);
    __syncthreads();
  }
  float mx = red[0]; __syncthreads();
  float e = __expf(s - mx);
  red[t] = e; __syncthreads();
  for (int off = 256; off > 0; off >>= 1) {
    if (t < off) red[t] += red[t + off];
    __syncthreads();
  }
  attn[b * KSEQ + t] = e * (1.f / red[0]);
}

__global__ void zero_kernel(float* __restrict__ p, int n) {
  int i = blockIdx.x * blockDim.x + threadIdx.x;
  if (i < n) p[i] = 0.f;
}

// g[b,h] = sum_k attn[b,k] * relu(x[b,k]@W_c1 + b_c1)[h]
__global__ void __launch_bounds__(256)
gsum_kernel(const float* __restrict__ x, const __bf16* __restrict__ Wc1T,
            const float* __restrict__ b_c1, const float* __restrict__ attn,
            float* __restrict__ g)
{
  __shared__ __bf16 sA[MTILE * LDPAD];
  __shared__ __bf16 sB[2][H_DIM * LDPAD];
  const int t = threadIdx.x, lane = t & 31, wave = t >> 5;
  const int m_sub = wave >> 2, n_group = wave & 3;
  const int m0 = blockIdx.x * MTILE;

  v8f acc[8];
  v8f z = {0.f,0.f,0.f,0.f,0.f,0.f,0.f,0.f};
#pragma unroll
  for (int j = 0; j < 8; ++j) acc[j] = z;

  gemm_tile_mainloop(x, Wc1T, m0, lane, m_sub, n_group, sA, sB[0], sB[1], acc);

  const int rh = (lane < 16) ? 0 : 8;
  const int b = m0 >> 9;                 // MTILE=32 divides KSEQ=512 -> whole tile in one batch
  float aw[8];
#pragma unroll
  for (int v = 0; v < 8; ++v) aw[v] = attn[m0 + m_sub * 16 + rh + v];
#pragma unroll
  for (int j = 0; j < 8; ++j) {
    int col = (n_group * 8 + j) * 16 + (lane & 15);
    float bias = b_c1[col];
    float p = 0.f;
#pragma unroll
    for (int v = 0; v < 8; ++v)
      p += fmaxf(acc[j][v] + bias, 0.f) * aw[v];
    p += __shfl_xor(p, 16);              // lanes L and L+16 share column
    if (lane < 16) atomicAdd(&g[b * H_DIM + col], p);
  }
}

// summed = g@W_c2 + b_c2 ; h = summed@W_ws + b_ws ; LN ; out = hn@W_u + b_u   (tiny: fp32 VALU)
__global__ void __launch_bounds__(256)
tail_kernel(const float* __restrict__ g, const float* __restrict__ W_c2, const float* __restrict__ b_c2,
            const float* __restrict__ W_ws, const float* __restrict__ b_ws,
            const float* __restrict__ ln_g, const float* __restrict__ ln_b,
            const float* __restrict__ W_u, const float* __restrict__ b_u,
            float* __restrict__ out)
{
  __shared__ float sg[H_DIM];
  __shared__ float ssum[D_DIM];
  __shared__ float sh[H_DIM];
  __shared__ float red[256];
  const int b = blockIdx.x, t = threadIdx.x;
  for (int i = t; i < H_DIM; i += 256) sg[i] = g[b * H_DIM + i];
  __syncthreads();
  for (int d = t; d < D_DIM; d += 256) {
    float acc = b_c2[d];
    for (int h = 0; h < H_DIM; ++h) acc += sg[h] * W_c2[h * D_DIM + d];
    ssum[d] = acc;
  }
  __syncthreads();
  for (int j = t; j < H_DIM; j += 256) {
    float acc = b_ws[j];
    for (int d = 0; d < D_DIM; ++d) acc += ssum[d] * W_ws[d * H_DIM + j];
    sh[j] = acc;
  }
  __syncthreads();
  float ps = 0.f, ps2 = 0.f;
  for (int j = t; j < H_DIM; j += 256) { float v = sh[j]; ps += v; ps2 += v * v; }
  red[t] = ps; __syncthreads();
  for (int off = 128; off > 0; off >>= 1) { if (t < off) red[t] += red[t + off]; __syncthreads(); }
  float mu = red[0] * (1.f / H_DIM); __syncthreads();
  red[t] = ps2; __syncthreads();
  for (int off = 128; off > 0; off >>= 1) { if (t < off) red[t] += red[t + off]; __syncthreads(); }
  float var = red[0] * (1.f / H_DIM) - mu * mu;
  float rs = rsqrtf(var + 1e-6f);
  __syncthreads();
  for (int j = t; j < H_DIM; j += 256) sh[j] = (sh[j] - mu) * rs * ln_g[j] + ln_b[j];
  __syncthreads();
  if (t < O_DIM) {
    float acc = b_u[t];
    for (int j = 0; j < H_DIM; ++j) acc += sh[j] * W_u[j * O_DIM + t];
    out[b * O_DIM + t] = acc;
  }
}

extern "C" void kernel_launch(void* const* d_in, const int* in_sizes, int n_in,
                              void* d_out, int out_size, void* d_ws, size_t ws_size,
                              hipStream_t stream) {
  (void)in_sizes; (void)n_in; (void)out_size; (void)ws_size;
  const float* x    = (const float*)d_in[0];
  const float* W_as = (const float*)d_in[1];
  const float* b_as = (const float*)d_in[2];
  const float* W_af = (const float*)d_in[3];
  const float* b_af = (const float*)d_in[4];
  const float* W_c1 = (const float*)d_in[5];
  const float* b_c1 = (const float*)d_in[6];
  const float* W_c2 = (const float*)d_in[7];
  const float* b_c2 = (const float*)d_in[8];
  const float* W_ws = (const float*)d_in[9];
  const float* b_ws = (const float*)d_in[10];
  const float* ln_g = (const float*)d_in[11];
  const float* ln_b = (const float*)d_in[12];
  const float* W_u  = (const float*)d_in[13];
  const float* b_u  = (const float*)d_in[14];
  float* out = (float*)d_out;

  char* ws = (char*)d_ws;
  __bf16* WasT   = (__bf16*)ws;                              // 1 MB
  __bf16* Wc1T   = (__bf16*)(ws + (1 << 20));                // 1 MB
  float*  scores = (float*) (ws + (2 << 20));                // 128 KB
  float*  attn   = (float*) (ws + (2 << 20) + (128 << 10));  // 128 KB
  float*  g      = (float*) (ws + (2 << 20) + (256 << 10));  // 128 KB

  prep_weights<<<(D_DIM * H_DIM) / 256, 256, 0, stream>>>(W_as, W_c1, WasT, Wc1T);
  scores_kernel<<<MROWS / MTILE, 256, 0, stream>>>(x, WasT, b_as, W_af, b_af, scores);
  softmax_kernel<<<BATCH, 512, 0, stream>>>(scores, attn);
  zero_kernel<<<(BATCH * H_DIM + 255) / 256, 256, 0, stream>>>(g, BATCH * H_DIM);
  gsum_kernel<<<MROWS / MTILE, 256, 0, stream>>>(x, Wc1T, b_c1, attn, g);
  tail_kernel<<<BATCH, 256, 0, stream>>>(g, W_c2, b_c2, W_ws, b_ws, ln_g, ln_b, W_u, b_u, out);
}
